// MultiHeadAttention_17257178595834
// MI455X (gfx1250) — matrix operand order repaired and checked by reading, not compiled, not run on hardware
//
#include <hip/hip_runtime.h>
#include <hip/hip_bf16.h>

// ---------------------------------------------------------------------------
// MHA forward for MI455X (gfx1250): bf16 WMMA + TDM (tensor_load_to_lds)
// staged weight tiles, flash attention with online softmax.
// ---------------------------------------------------------------------------

typedef __attribute__((ext_vector_type(16))) __bf16 v16bf;
typedef __attribute__((ext_vector_type(8)))  __bf16 v8bf;
typedef __attribute__((ext_vector_type(8)))  float  v8f;
typedef __attribute__((ext_vector_type(4)))  unsigned int v4u;
typedef __attribute__((ext_vector_type(8)))  int v8i;
typedef __attribute__((ext_vector_type(4)))  int v4i;

#define BATCH  4
#define SEQ    2048
#define DMODEL 1024
#define NHEADS 16
#define DHEAD  64
#define MTOT   (BATCH * SEQ)     // 8192 tokens
#define LOG2E  1.44269504088896340736f

// ---- WMMA helper -----------------------------------------------------------
__device__ __forceinline__ v8f wmma_bf16(v16bf a, v16bf b, v8f c) {
  // D = A(16x32 bf16) * B(32x16 bf16) + C(16x16 f32)
  return __builtin_amdgcn_wmma_f32_16x16x32_bf16(
      /*neg_a=*/false, a, /*neg_b=*/false, b,
      /*c_mod=*/(short)0, c, /*reuse_a=*/false, /*reuse_b=*/false);
}

// A-fragment loader (16-bit A 16x32, ISA 7.12.2):
//   lane<16 : M=lane,    halves 0-7 = K k0+0..7,  halves 8-15 = K k0+16..23
//   lane>=16: M=lane-16, same +8
// Caller passes row pointer (M = lane&15) and kstart = k0 + (lane>>4)*8.
__device__ __forceinline__ v16bf load_a_frag(const __bf16* row, int kstart) {
  union { v16bf v; v8bf h[2]; } u;
  u.h[0] = *reinterpret_cast<const v8bf*>(row + kstart);
  u.h[1] = *reinterpret_cast<const v8bf*>(row + kstart + 16);
  return u.v;
}

// ---- Tensor Data Mover: async 2D tile load (global -> LDS) -----------------
// D# per CDNA5 ISA sect. 8: 2D tile, data_size = 2 bytes,
// tile_dim0 x tile_dim1 = 64 x 32 elements, row stride = DMODEL elements.
// Tensor dims set huge so no OOB clipping (tiles are always fully in-bounds).
// This toolchain exposes the 6-arg builtin:
//   (uint32x4 g0, int32x8 g1, int32x4, int32x4, int32x8, i32 cpol)
__device__ __forceinline__ void tdm_load_tile(const __bf16* gsrc, unsigned lds_off) {
  const unsigned long long ga = (unsigned long long)(uintptr_t)gsrc;
  v4u g0;
  g0[0] = 1u;                                             // count=1 (valid, user mode)
  g0[1] = lds_off;                                        // lds_addr (bytes)
  g0[2] = (unsigned)ga;                                   // global_addr[31:0]
  g0[3] = (unsigned)((ga >> 32) & 0x01ffffffu)            // global_addr[56:32]
          | 0x80000000u;                                  // type=2 ("image")
  const unsigned td0 = 1u << 30;                          // tensor_dim0 (huge)
  const unsigned td1 = 1u << 30;                          // tensor_dim1 (huge)
  const unsigned long long st0 = (unsigned long long)DMODEL; // dim0 stride (elems)
  v8i g1;
  g1[0] = 0x10000;                                        // data_size=1 -> 2 bytes
  g1[1] = (int)((td0 & 0xffffu) << 16);                   // tensor_dim0[15:0]
  g1[2] = (int)(((td0 >> 16) & 0xffffu) | ((td1 & 0xffffu) << 16));
  g1[3] = (int)(((td1 >> 16) & 0xffffu) | (64u << 16));   // tile_dim0 = 64
  g1[4] = (int)32u;                                       // tile_dim1 = 32, tile_dim2 = 0
  g1[5] = (int)(unsigned)(st0 & 0xffffffffu);             // dim0_stride lo
  g1[6] = (int)(unsigned)((st0 >> 32) & 0xffffu);         // dim0_stride hi, dim1_stride lo16=0
  g1[7] = 0;
  const v4i z4 = {0, 0, 0, 0};                            // groups 2/3 unused (2D)
  const v8i z8 = {0, 0, 0, 0, 0, 0, 0, 0};
  __builtin_amdgcn_tensor_load_to_lds(g0, g1, z4, z4, z8, 0);
}

// ---- elementwise converters ------------------------------------------------
__global__ __launch_bounds__(256) void cvt_f32_bf16(const float* __restrict__ in,
                                                    __bf16* __restrict__ out, int n) {
  int i = blockIdx.x * blockDim.x + threadIdx.x;
  if (i < n) out[i] = (__bf16)in[i];
}

// Wt[k][n] = (bf16) W[n][k]   (W is [DMODEL][DMODEL] row-major, y = x @ W^T)
__global__ __launch_bounds__(256) void cvt_transpose_bf16(const float* __restrict__ W,
                                                          __bf16* __restrict__ Wt) {
  int i = blockIdx.x * blockDim.x + threadIdx.x;          // DMODEL*DMODEL threads
  int k = i / DMODEL;
  int n = i % DMODEL;                                     // coalesced writes
  Wt[(size_t)k * DMODEL + n] = (__bf16)W[(size_t)n * DMODEL + k];
}

// ---- WMMA GEMM with TDM-staged weights:  C[8192,1024] = A @ Bw[K,N] --------
// Block (8 waves) computes a 256x64 tile; each wave a 32x64 slice (8 wmma/step).
// B tile (32x64 bf16, 4KB) is DMA'd into LDS by the Tensor Data Mover, double
// buffered, shared by all 8 waves.  A fragments stream from global (L2).
// mode 0: bf16 -> Q [b,h,s,d] * oscale   mode 1: bf16 -> Kt [b,h,d,s]
// mode 2: bf16 -> V [b,h,s,d]            mode 3: f32  -> plain [M,N]
__global__ __launch_bounds__(256) void gemm_wmma_tdm(
    const __bf16* __restrict__ A, const __bf16* __restrict__ Bw,
    float* __restrict__ outF, __bf16* __restrict__ outB,
    int mode, float oscale)
{
  __shared__ alignas(16) __bf16 bbuf[2][32][64];          // 2 x 4KB B tiles

  const int lane = threadIdx.x & 31;
  const int li   = lane & 15;
  const int hf   = lane >> 4;
  const int wv   = threadIdx.x >> 5;                      // wave in block (0..7)
  const int NT   = DMODEL / 64;                           // 16 n-tiles
  const int mt   = blockIdx.x / NT;                       // 0..31 (256-row tiles)
  const int nt   = blockIdx.x % NT;
  const int m0   = mt * 256 + wv * 32;                    // wave's 32-row slice
  const int n0   = nt * 64;
  const bool issuer = (wv == 0);

  v8f acc[2][4] = {};
  const __bf16* arow0 = A + (size_t)(m0 + li) * DMODEL;
  const __bf16* arow1 = A + (size_t)(m0 + 16 + li) * DMODEL;
  const int kb = hf * 8;

  if (issuer)
    tdm_load_tile(Bw + n0, (unsigned)(uintptr_t)&bbuf[0][0][0]);

  for (int ks = 0; ks < DMODEL / 32; ++ks) {
    const int k = ks * 32;
    if (issuer) __builtin_amdgcn_s_wait_tensorcnt(0);     // tile ks landed in LDS
    __syncthreads();                                      // publish to all waves
    if (issuer && (ks + 1) < DMODEL / 32)                 // prefetch next tile
      tdm_load_tile(Bw + (size_t)(k + 32) * DMODEL + n0,
                    (unsigned)(uintptr_t)&bbuf[(ks + 1) & 1][0][0]);

    v16bf a0 = load_a_frag(arow0, k + kb);
    v16bf a1 = load_a_frag(arow1, k + kb);
#pragma unroll
    for (int t = 0; t < 4; ++t) {
      v16bf b;
#pragma unroll
      for (int s = 0; s < 16; ++s)
        b[s] = bbuf[ks & 1][(s & 7) + 8 * hf + ((s >> 3) << 4)][t * 16 + li];
      acc[0][t] = wmma_bf16(a0, b, acc[0][t]);
      acc[1][t] = wmma_bf16(a1, b, acc[1][t]);
    }
  }

  // C/D layout: VGPR j holds row (j + 8*hf), col = li
#pragma unroll
  for (int sub = 0; sub < 2; ++sub) {
#pragma unroll
    for (int t = 0; t < 4; ++t) {
#pragma unroll
      for (int j = 0; j < 8; ++j) {
        const int r = m0 + sub * 16 + hf * 8 + j;         // token index
        const int c = n0 + t * 16 + li;                   // feature index
        const float v = acc[sub][t][j] * oscale;
        if (mode == 3) {
          outF[(size_t)r * DMODEL + c] = v;
        } else {
          const int bb = r / SEQ, ss = r % SEQ;
          const int hh = c / DHEAD, dd = c % DHEAD;
          size_t idx;
          if (mode == 1)
            idx = ((size_t)(bb * NHEADS + hh) * DHEAD + dd) * SEQ + ss;  // [b,h,d,s]
          else
            idx = ((size_t)(bb * NHEADS + hh) * SEQ + ss) * DHEAD + dd;  // [b,h,s,d]
          outB[idx] = (__bf16)v;
        }
      }
    }
  }
}

// ---- flash attention (causal): one wave per 16-row q tile ------------------
// Q: [b,h,s,d] bf16 (pre-scaled by 1/sqrt(Dh));  Kt: [b,h,d,s];  V: [b,h,s,d]
// O: [b, s, h*Dh] bf16 (ready as A operand of the output projection)
__global__ __launch_bounds__(256) void attn_wmma(
    const __bf16* __restrict__ Q, const __bf16* __restrict__ Kt,
    const __bf16* __restrict__ V, __bf16* __restrict__ O)
{
  __shared__ alignas(16) __bf16 lds[8][16][32];           // per-wave P staging

  const int lane  = threadIdx.x & 31;
  const int li    = lane & 15;
  const int hf    = lane >> 4;
  const int wslot = threadIdx.x >> 5;
  const int wid   = blockIdx.x * (blockDim.x >> 5) + wslot;
  const int QT    = SEQ / 16;                             // 128 q-tiles per (b,h)
  if (wid >= BATCH * NHEADS * QT) return;
  const int qt = wid % QT;
  const int bh = wid / QT;
  const int hd = bh % NHEADS;
  const int bb = bh / NHEADS;
  const int q0 = qt * 16;

  const __bf16* Qh = Q  + ((size_t)bh * SEQ + q0) * DHEAD;
  const __bf16* Kh = Kt + (size_t)bh * DHEAD * SEQ;
  const __bf16* Vh = V  + (size_t)bh * SEQ * DHEAD;

  v16bf aQ[2];
#pragma unroll
  for (int f = 0; f < 2; ++f)
    aQ[f] = load_a_frag(Qh + (size_t)li * DHEAD, f * 32 + hf * 8);

  float m_s[8], l_s[8];
#pragma unroll
  for (int j = 0; j < 8; ++j) { m_s[j] = -3.0e38f; l_s[j] = 0.0f; }
  v8f acc[4] = {};

  const int kend = q0 + 16;
  for (int kc = 0; kc < kend; kc += 32) {
    v8f sc[2];
#pragma unroll
    for (int t = 0; t < 2; ++t) {
      v8f c = {};
      const int k0 = kc + 16 * t;
#pragma unroll
      for (int f = 0; f < 2; ++f) {
        v16bf bK;
#pragma unroll
        for (int s = 0; s < 16; ++s)
          bK[s] = Kh[(size_t)(f * 32 + (s & 7) + 8 * hf + ((s >> 3) << 4)) * SEQ + k0 + li];
        c = wmma_bf16(aQ[f], bK, c);
      }
      sc[t] = c;
    }
    if (kc + 31 > q0) {                                   // causal mask near diag
#pragma unroll
      for (int t = 0; t < 2; ++t) {
        const int kcol = kc + 16 * t + li;
#pragma unroll
        for (int j = 0; j < 8; ++j) {
          const int qrow = q0 + hf * 8 + j;
          if (kcol > qrow) sc[t][j] = -1.0e9f;
        }
      }
    }
    float rmx[8];
#pragma unroll
    for (int j = 0; j < 8; ++j) rmx[j] = fmaxf(sc[0][j], sc[1][j]);
#pragma unroll
    for (int d = 1; d < 16; d <<= 1) {
#pragma unroll
      for (int j = 0; j < 8; ++j)
        rmx[j] = fmaxf(rmx[j], __shfl_xor(rmx[j], d, 32));
    }
    float psc[8];
#pragma unroll
    for (int j = 0; j < 8; ++j) {
      const float mn = fmaxf(m_s[j], rmx[j]);
      psc[j] = exp2f((m_s[j] - mn) * LOG2E);
      m_s[j] = mn;
    }
#pragma unroll
    for (int t = 0; t < 4; ++t)
#pragma unroll
      for (int j = 0; j < 8; ++j) acc[t][j] *= psc[j];
    float rsum[8];
#pragma unroll
    for (int j = 0; j < 8; ++j) rsum[j] = 0.0f;
#pragma unroll
    for (int t = 0; t < 2; ++t) {
#pragma unroll
      for (int j = 0; j < 8; ++j) {
        const float p = exp2f((sc[t][j] - m_s[j]) * LOG2E);
        sc[t][j] = p;
        rsum[j] += p;
      }
    }
#pragma unroll
    for (int d = 1; d < 16; d <<= 1) {
#pragma unroll
      for (int j = 0; j < 8; ++j) rsum[j] += __shfl_xor(rsum[j], d, 32);
    }
#pragma unroll
    for (int j = 0; j < 8; ++j) l_s[j] = l_s[j] * psc[j] + rsum[j];

    // re-layout P: C-fragment -> LDS -> A-fragment (16x32 bf16, wave-private)
#pragma unroll
    for (int t = 0; t < 2; ++t)
#pragma unroll
      for (int j = 0; j < 8; ++j)
        lds[wslot][hf * 8 + j][t * 16 + li] = (__bf16)sc[t][j];
    v16bf aP;
    {
      union { v16bf v; v8bf h[2]; } u;
      u.h[0] = *reinterpret_cast<const v8bf*>(&lds[wslot][li][hf * 8]);
      u.h[1] = *reinterpret_cast<const v8bf*>(&lds[wslot][li][hf * 8 + 16]);
      aP = u.v;
    }
#pragma unroll
    for (int t = 0; t < 4; ++t) {
      v16bf bV;
#pragma unroll
      for (int s = 0; s < 16; ++s)
        bV[s] = Vh[(size_t)(kc + (s & 7) + 8 * hf + ((s >> 3) << 4)) * DHEAD + t * 16 + li];
      acc[t] = wmma_bf16(aP, bV, acc[t]);
    }
  }

  float inv[8];
#pragma unroll
  for (int j = 0; j < 8; ++j) inv[j] = 1.0f / l_s[j];
#pragma unroll
  for (int t = 0; t < 4; ++t) {
#pragma unroll
    for (int j = 0; j < 8; ++j) {
      const float v = acc[t][j] * inv[j];
      const int srow = q0 + hf * 8 + j;
      O[((size_t)bb * SEQ + srow) * DMODEL + hd * DHEAD + t * 16 + li] = (__bf16)v;
    }
  }
}

// ---------------------------------------------------------------------------
extern "C" void kernel_launch(void* const* d_in, const int* in_sizes, int n_in,
                              void* d_out, int out_size, void* d_ws, size_t ws_size,
                              hipStream_t stream) {
  const float* x  = (const float*)d_in[0];
  const float* Wq = (const float*)d_in[1];
  const float* Wk = (const float*)d_in[2];
  const float* Wv = (const float*)d_in[3];
  const float* Wo = (const float*)d_in[4];
  float* out = (float*)d_out;

  char* ws = (char*)d_ws;
  const size_t SZ_XB  = (size_t)MTOT * DMODEL * sizeof(__bf16);                  // 16 MB
  const size_t SZ_W   = (size_t)DMODEL * DMODEL * sizeof(__bf16);                //  2 MB
  const size_t SZ_QKV = (size_t)BATCH * NHEADS * SEQ * DHEAD * sizeof(__bf16);   // 16 MB

  __bf16* xb    = (__bf16*)(ws);                ws += SZ_XB;
  __bf16* Wqt   = (__bf16*)(ws);                ws += SZ_W;
  __bf16* Wkt   = (__bf16*)(ws);                ws += SZ_W;
  __bf16* Wvt   = (__bf16*)(ws);                ws += SZ_W;
  __bf16* Wot   = (__bf16*)(ws);                ws += SZ_W;
  __bf16* Qb    = (__bf16*)(ws);                ws += SZ_QKV;
  __bf16* Ktb   = (__bf16*)(ws);                ws += SZ_QKV;
  __bf16* Vb    = (__bf16*)(ws);                ws += SZ_QKV;
  __bf16* attnb = (__bf16*)(ws);                ws += SZ_XB;

  // 1) precision conversion + weight transposes
  cvt_f32_bf16<<<(MTOT * DMODEL) / 256, 256, 0, stream>>>(x, xb, MTOT * DMODEL);
  cvt_transpose_bf16<<<(DMODEL * DMODEL) / 256, 256, 0, stream>>>(Wq, Wqt);
  cvt_transpose_bf16<<<(DMODEL * DMODEL) / 256, 256, 0, stream>>>(Wk, Wkt);
  cvt_transpose_bf16<<<(DMODEL * DMODEL) / 256, 256, 0, stream>>>(Wv, Wvt);
  cvt_transpose_bf16<<<(DMODEL * DMODEL) / 256, 256, 0, stream>>>(Wo, Wot);

  // 2) QKV projections: 256x64 block tiles -> 512 blocks of 8 waves
  const int gemm_blocks = (MTOT / 256) * (DMODEL / 64);   // 512
  gemm_wmma_tdm<<<gemm_blocks, 256, 0, stream>>>(xb, Wqt, nullptr, Qb, 0, 0.125f);
  gemm_wmma_tdm<<<gemm_blocks, 256, 0, stream>>>(xb, Wkt, nullptr, Ktb, 1, 1.0f);
  gemm_wmma_tdm<<<gemm_blocks, 256, 0, stream>>>(xb, Wvt, nullptr, Vb, 2, 1.0f);

  // 3) causal flash attention (8192 q-tiles -> 1024 blocks of 8 waves)
  const int attn_blocks = (BATCH * NHEADS * (SEQ / 16)) / 8;   // 1024
  attn_wmma<<<attn_blocks, 256, 0, stream>>>(Qb, Ktb, Vb, attnb);

  // 4) output projection -> f32 d_out
  gemm_wmma_tdm<<<gemm_blocks, 256, 0, stream>>>(attnb, Wot, out, nullptr, 3, 1.0f);
}
